// SupConLoss_20315195310275
// MI455X (gfx1250) — matrix operand order repaired and checked by reading, not compiled, not run on hardware
//
#include <hip/hip_runtime.h>
#include <hip/hip_bf16.h>

typedef __attribute__((ext_vector_type(16))) _Float16 v16h;
typedef __attribute__((ext_vector_type(8)))  _Float16 v8h;
typedef __attribute__((ext_vector_type(8)))  float    v8f;

#define DCONST 128
#define INV_T  14.285714285714286f     // 1 / 0.07

#define ROW_BYTES   256                // 128 f16
#define LDS_STRIDE  272                // 256 + 16 pad: kills 256B-stride bank conflicts
#define TILE_BYTES  (16 * LDS_STRIDE)  // one 16-row staged tile
#define TILES_STEP  4                  // column tiles staged per step
#define STRIPS      2                  // row strips per WG  -> 32 rows / WG
#define NSTEPS(N)   ((N) / (16 * TILES_STEP))

union V16U { v16h v; v8h p[2]; };

// ---- CDNA5 async global->LDS copy (ASYNCcnt-tracked), 16 bytes per lane ----
__device__ __forceinline__ void async_ld_b128(unsigned lds_byte_addr,
                                              const void* gaddr) {
  asm volatile("global_load_async_to_lds_b128 %0, %1, off"
               :: "v"(lds_byte_addr), "v"((unsigned long long)(uintptr_t)gaddr)
               : "memory");
}
__device__ __forceinline__ void wait_async0() {
  asm volatile("s_wait_asynccnt 0" ::: "memory");
}

// ---------------------------------------------------------------------------
__global__ void zero_out_kernel(float* out) { out[0] = 0.0f; }

// ---------------------------------------------------------------------------
// Kernel 1: L2-normalize each contrast row, cast to f16, contrast ordering:
// row i <- features[(i % B), (i / B), :].  One 128-thread block per row.
// ---------------------------------------------------------------------------
__global__ __launch_bounds__(128)
void normalize_kernel(const float* __restrict__ feat,
                      _Float16* __restrict__ zn,
                      int B, int N) {
  const int i = blockIdx.x;
  const int d = threadIdx.x;
  const int b = i % B;
  const int v = i / B;

  const float x = feat[((size_t)b * 2 + v) * DCONST + d];

  float s = x * x;
  #pragma unroll
  for (int off = 16; off > 0; off >>= 1) s += __shfl_xor(s, off, 32);

  __shared__ float wsum[4];
  const int lane = threadIdx.x & 31;
  const int wid  = threadIdx.x >> 5;
  if (lane == 0) wsum[wid] = s;
  __syncthreads();
  const float tot = wsum[0] + wsum[1] + wsum[2] + wsum[3];
  const float nrm = fmaxf(sqrtf(tot), 1e-8f);

  zn[(size_t)i * DCONST + d] = (_Float16)(x / nrm);
}

// ---------------------------------------------------------------------------
// Kernel 2: WG = 32 rows (2 strips x 4 column-parity waves). Per step, stage
// 4 column tiles into LDS with double-buffered global_load_async_to_lds_b128,
// then each wave runs 4x v_wmma_f32_16x16x32_f16 (K=128) against its hoisted
// A operand, followed by branchless diagonal masking / exp accumulation and
// positive-logit capture. Wave -> LDS -> scalar atomic reduction.
// ---------------------------------------------------------------------------
__global__ __launch_bounds__(256)
void supcon_tiles_kernel(const _Float16* __restrict__ zn,
                         float* __restrict__ out,
                         int B, int N) {
  const int tid    = threadIdx.x;
  const int lane   = tid & 31;
  const int wave   = tid >> 5;           // 0..7
  const int nIdx   = lane & 15;
  const int h      = lane >> 4;
  const int strip  = wave >> 2;          // 0..1
  const int parity = wave & 3;           // 0..3
  const int rbase  = blockIdx.x * (16 * STRIPS) + strip * 16;

  __shared__ char  stage[2][TILES_STEP][TILE_BYTES];
  __shared__ float red_sum[16 * STRIPS];
  __shared__ float red_pos[16 * STRIPS];

  // ---- hoisted A operand: this wave's 16 rows, 4 chunks of K=32 -----------
  V16U a[4];
  {
    const _Float16* rp = zn + (size_t)(rbase + nIdx) * DCONST;
    #pragma unroll
    for (int c = 0; c < 4; ++c) {
      a[c].p[0] = *(const v8h*)(rp + 32 * c + 8 * h);       // K {0-7 | 8-15}
      a[c].p[1] = *(const v8h*)(rp + 32 * c + 16 + 8 * h);  // K {16-23 | 24-31}
    }
  }

  if (tid < 16 * STRIPS) { red_sum[tid] = 0.0f; red_pos[tid] = 0.0f; }

  float sumacc[8], posacc[8];
  #pragma unroll
  for (int r = 0; r < 8; ++r) { sumacc[r] = 0.0f; posacc[r] = 0.0f; }

  // staging geometry: thread t copies chunk (tile q, row t>>4, 16B piece t&15)
  const int srow  = tid >> 4;            // 0..15
  const int schk  = tid & 15;            // 0..15
  const unsigned lds_elem = (unsigned)(srow * LDS_STRIDE + schk * 16);

  const int nsteps = NSTEPS(N);          // 128 for N=8192

  // ---- prime buffer 0 -----------------------------------------------------
  {
    const int jb0 = 0;                   // step 0: tiles 0..3
    #pragma unroll
    for (int q = 0; q < TILES_STEP; ++q) {
      unsigned ldsa = (unsigned)(uintptr_t)&stage[0][q][0] + lds_elem;
      const _Float16* g = zn + (size_t)(jb0 + q * 16 + srow) * DCONST + schk * 8;
      async_ld_b128(ldsa, g);
    }
  }

  for (int s = 0; s < nsteps; ++s) {
    const int cur = s & 1;

    wait_async0();        // this wave's staged loads for buf[cur] are in LDS
    __syncthreads();      // all waves staged; all done reading buf[cur^1]

    // ---- prefetch next step into the other buffer (overlaps compute) ------
    if (s + 1 < nsteps) {
      const int jb = (s + 1) * (16 * TILES_STEP);
      #pragma unroll
      for (int q = 0; q < TILES_STEP; ++q) {
        unsigned ldsa = (unsigned)(uintptr_t)&stage[cur ^ 1][q][0] + lds_elem;
        const _Float16* g = zn + (size_t)(jb + q * 16 + srow) * DCONST + schk * 8;
        async_ld_b128(ldsa, g);
      }
    }

    // ---- B operand from LDS (padded rows -> conflict-light ds_load_b128) --
    const int jbase = s * (16 * TILES_STEP) + parity * 16;
    const char* tp = &stage[cur][parity][0] + nIdx * LDS_STRIDE;
    V16U bm[4];
    #pragma unroll
    for (int c = 0; c < 4; ++c) {
      bm[c].p[0] = *(const v8h*)(tp + 64 * c + 16 * h);
      bm[c].p[1] = *(const v8h*)(tp + 64 * c + 16 * h + 32);
    }

    // ---- 16x16 logits tile via 4 chained WMMAs (K=128) --------------------
    v8f acc = {};
    #pragma unroll
    for (int c = 0; c < 4; ++c) {
      acc = __builtin_amdgcn_wmma_f32_16x16x32_f16(
          false, a[c].v, false, bm[c].v, (short)0, acc, false, false);
    }

    // ---- branchless masking + accumulation --------------------------------
    #pragma unroll
    for (int r = 0; r < 8; ++r) {
      const int gi    = rbase + r + 8 * h;
      const int gj    = jbase + nIdx;
      const float lgt = acc[r] * INV_T;
      const float e   = (gi == gj) ? 0.0f : __expf(lgt);
      sumacc[r] += e;
      const int pos = (gi < B) ? (gi + B) : (gi - B);
      posacc[r] += (gj == pos) ? lgt : 0.0f;
    }

    __syncthreads();      // everyone done reading buf[cur] before it refills
  }

  // ---- reduce 16 column slots per lane half -------------------------------
  #pragma unroll
  for (int r = 0; r < 8; ++r) {
    #pragma unroll
    for (int off = 8; off > 0; off >>= 1) {
      sumacc[r] += __shfl_xor(sumacc[r], off, 16);
      posacc[r] += __shfl_xor(posacc[r], off, 16);
    }
  }

  if ((lane & 15) == 0) {
    #pragma unroll
    for (int r = 0; r < 8; ++r) {
      const int li = strip * 16 + r + 8 * h;
      atomicAdd(&red_sum[li], sumacc[r]);
      atomicAdd(&red_pos[li], posacc[r]);
    }
  }
  __syncthreads();

  if (tid < 16 * STRIPS) {
    const float v = logf(red_sum[tid]) - red_pos[tid];
    atomicAdd(out, v * (1.0f / (float)N));
  }
}

// ---------------------------------------------------------------------------
extern "C" void kernel_launch(void* const* d_in, const int* in_sizes, int n_in,
                              void* d_out, int out_size, void* d_ws, size_t ws_size,
                              hipStream_t stream) {
  const float* feat = (const float*)d_in[0];
  const int total = in_sizes[0];          // B * 2 * 128
  const int B = total / (2 * DCONST);     // 4096
  const int N = 2 * B;                    // 8192

  _Float16* zn = (_Float16*)d_ws;         // N * 128 f16 = 2 MB scratch
  float* out = (float*)d_out;

  hipLaunchKernelGGL(zero_out_kernel, dim3(1), dim3(1), 0, stream, out);
  hipLaunchKernelGGL(normalize_kernel, dim3(N), dim3(128), 0, stream,
                     feat, zn, B, N);
  hipLaunchKernelGGL(supcon_tiles_kernel, dim3(N / (16 * STRIPS)), dim3(256),
                     0, stream, zn, out, B, N);
}
